// SingleLSTM_10101763080399
// MI455X (gfx1250) — matrix operand order repaired and checked by reading, not compiled, not run on hardware
//
#include <hip/hip_runtime.h>
#include <hip/hip_bf16.h>
#include <stdint.h>
#include <stddef.h>

// ---------------------------------------------------------------------------
// Problem constants (from reference): B=2048, OBS=8, PRED=12, H=512, L=2
// ---------------------------------------------------------------------------
#define BSZ   2048
#define OBSN  8
#define PREDN 12
#define HID   512
#define GSZ   2048      // 4*H gates
#define KTOT  1024      // [x(512) | h(512)] combined K
#define NL    2

typedef __bf16 bf16;
typedef __bf16 v16bf __attribute__((ext_vector_type(16)));
typedef __bf16 v8bf  __attribute__((ext_vector_type(8)));
typedef float  v8f   __attribute__((ext_vector_type(8)));
typedef int    v4i   __attribute__((ext_vector_type(4)));

#define AS1 __attribute__((address_space(1)))
#define AS3 __attribute__((address_space(3)))

#if defined(__AMDGCN__) && __has_builtin(__builtin_amdgcn_global_load_async_to_lds_b128)
#define USE_ASYNC_LDS 1
#else
#define USE_ASYNC_LDS 0
#endif

__device__ __forceinline__ bf16 f2bf(float f) {
    uint32_t u = __builtin_bit_cast(uint32_t, f);
    uint32_t r = (u + 0x7FFFu + ((u >> 16) & 1u)) >> 16;
    return __builtin_bit_cast(bf16, (uint16_t)r);
}
__device__ __forceinline__ float sigm(float x) { return 1.0f / (1.0f + __expf(-x)); }

__device__ __forceinline__ void wait_async_all() {
#if USE_ASYNC_LDS
#if __has_builtin(__builtin_amdgcn_s_wait_asynccnt)
    __builtin_amdgcn_s_wait_asynccnt(0);
#else
    asm volatile("s_wait_asynccnt 0" ::: "memory");
#endif
#endif
}

// ---------------------------------------------------------------------------
// One-time prep kernels
// ---------------------------------------------------------------------------

// Pack combined [Wih | Whh] (f32, (L,4H,H) each) into bf16 B-fragment layout:
// per (layer, ntile=(n/16), kblk=(k/32)): 32 lanes x 16 bf16 contiguous.
//   lane = (k%32 < 16) ? n%16 : 16 + n%16 ; elem = k%16
__global__ __launch_bounds__(256) void pack_w_kernel(const float* __restrict__ Wih,
                                                     const float* __restrict__ Whh,
                                                     bf16* __restrict__ wpk) {
    int idx = blockIdx.x * blockDim.x + threadIdx.x;
    if (idx >= NL * GSZ * KTOT) return;
    int l = idx / (GSZ * KTOT);
    int rem = idx - l * (GSZ * KTOT);
    int n = rem / KTOT;
    int k = rem - n * KTOT;
    float v = (k < HID) ? Wih[((size_t)l * GSZ + n) * HID + k]
                        : Whh[((size_t)l * GSZ + n) * HID + (k - HID)];
    int ntile = n >> 4, n16 = n & 15;
    int kblk = k >> 5, kin = k & 31;
    int lane = (kin < 16) ? n16 : (n16 + 16);
    int e = kin & 15;
    size_t dst = (size_t)l * (GSZ * KTOT) +
                 (((size_t)ntile * 32 + kblk) * 32 + lane) * 16 + e;
    wpk[dst] = f2bf(v);
}

__global__ void bias_kernel(const float* __restrict__ bih, const float* __restrict__ bhh,
                            float* __restrict__ bias) {
    int idx = blockIdx.x * blockDim.x + threadIdx.x;
    if (idx >= NL * GSZ) return;
    bias[idx] = bih[idx] + bhh[idx];
}

// embed[t][b][j] = ((obsVel-mean)/std) @ enc_W + enc_b  (t = 0..6), stored bf16
__global__ __launch_bounds__(256) void embed_kernel(const float* __restrict__ obsVel,
                                                    const float* __restrict__ mean,
                                                    const float* __restrict__ stdv,
                                                    const float* __restrict__ encW,
                                                    const float* __restrict__ encb,
                                                    bf16* __restrict__ embed) {
    int idx = blockIdx.x * blockDim.x + threadIdx.x;
    if (idx >= 7 * BSZ * HID) return;
    int t = idx / (BSZ * HID);
    int rem = idx - t * (BSZ * HID);
    int b = rem / HID;
    int j = rem - b * HID;
    float x0 = (obsVel[((size_t)b * OBSN + t) * 2 + 0] - mean[0]) / stdv[0];
    float x1 = (obsVel[((size_t)b * OBSN + t) * 2 + 1] - mean[1]) / stdv[1];
    float e = fmaf(x0, encW[j], fmaf(x1, encW[HID + j], encb[j]));
    embed[idx] = f2bf(e);
}

__global__ void zero_u32_kernel(uint32_t* __restrict__ p, int n) {
    int idx = blockIdx.x * blockDim.x + threadIdx.x;
    if (idx < n) p[idx] = 0u;
}

// Copy embed slab t into layer-0 activation x-part (cols 0..511 of KTOT row)
__global__ void set_x0_kernel(const bf16* __restrict__ embed_t, bf16* __restrict__ act0) {
    int idx = blockIdx.x * blockDim.x + threadIdx.x;
    if (idx >= BSZ * HID) return;
    int b = idx / HID;
    int j = idx - b * HID;
    act0[(size_t)b * KTOT + j] = embed_t[idx];
}

// ---------------------------------------------------------------------------
// WMMA GEMM:  gates(2048 x 2048) = act(2048 x 1024) * Wpk^T + bias
// block = 256 threads = 8 waves (2 M x 4 N), block tile 128x128.
// A tile (128x32 bf16) staged to LDS per K-step via global_load_async_to_lds,
// double-buffered (ASYNCcnt). Row stride padded to 80B (bank-conflict-free).
// B fragments direct from packed global (per-lane contiguous, L2-resident).
// ---------------------------------------------------------------------------
#define A_ROWB   80          // padded LDS row stride (bytes) for a 64B row
#define A_TILEB  (128 * A_ROWB)   // 10240 bytes per buffer

__global__ __launch_bounds__(256) void gemm_gates_kernel(const bf16* __restrict__ act,
                                                         const bf16* __restrict__ wpk,
                                                         const float* __restrict__ bias,
                                                         float* __restrict__ gates) {
    __shared__ __align__(16) char smem[2 * A_TILEB];

    const int lane = threadIdx.x & 31;
    const int wave = threadIdx.x >> 5;
    const int wm = wave >> 2;              // 0..1
    const int wn = wave & 3;               // 0..3
    const int blockM = blockIdx.y * 128;
    const int m0w = wm * 64;               // wave M base within block
    const int n0 = blockIdx.x * 128 + wn * 32;

    const int lrow  = lane & 15;
    const int khalf = lane >> 4;           // 0 or 1 (A-fragment K interleave)

    v8f acc[4][2] = {};

    const int ntile0 = n0 >> 4;
    // Per-lane packed-B base: ((ntile*32 + kblk)*32 + lane)*16 bf16
    const bf16* bbase0 = wpk + (((size_t)ntile0 * 32) * 32 + lane) * 16;
    const bf16* bbase1 = wpk + (((size_t)(ntile0 + 1) * 32) * 32 + lane) * 16;

    // Stage A tile for K-block kb into LDS buffer buf (2 x 16B chunks / thread)
    auto issueA = [&](int kb, int buf) {
#pragma unroll
        for (int i = 0; i < 2; ++i) {
            const int c = threadIdx.x + i * 256;       // chunk id 0..511
            const int row = c >> 2;                    // 0..127
            const int sub = c & 3;                     // 16B chunk within 64B row
            const bf16* gp = act + (size_t)(blockM + row) * KTOT + kb * 32 + sub * 8;
            char* lp = smem + buf * A_TILEB + row * A_ROWB + sub * 16;
#if USE_ASYNC_LDS
            __builtin_amdgcn_global_load_async_to_lds_b128(
                (AS1 v4i*)gp, (AS3 v4i*)lp, 0, 0);
#else
            *(v8bf*)lp = *(const v8bf*)gp;
#endif
        }
    };

    issueA(0, 0);
    wait_async_all();
    __syncthreads();

    for (int kb = 0; kb < 32; ++kb) {
        const int cur = kb & 1;
        if (kb + 1 < 32) issueA(kb + 1, cur ^ 1);      // overlap next tile with compute

        // B fragments: 32 contiguous bytes per lane from packed weights
        v16bf bfrag0 = *(const v16bf*)(bbase0 + (size_t)kb * (32 * 16));
        v16bf bfrag1 = *(const v16bf*)(bbase1 + (size_t)kb * (32 * 16));
        if (kb + 1 < 32) { // warm L2/L0 for next K-block's weights
            __builtin_prefetch(bbase0 + (size_t)(kb + 1) * (32 * 16), 0, 1);
            __builtin_prefetch(bbase1 + (size_t)(kb + 1) * (32 * 16), 0, 1);
        }

        // A fragments from LDS: lane<16 holds row m, K=[0..15] of block;
        //                       lane>=16 holds K=[16..31]
        const char* abase = smem + cur * A_TILEB;
        v16bf afrag[4];
#pragma unroll
        for (int mt = 0; mt < 4; ++mt) {
            const char* p = abase + (m0w + mt * 16 + lrow) * A_ROWB + khalf * 32;
            union { v16bf v; struct { v8bf lo; v8bf hi; } s; } u;
            u.s.lo = *(const v8bf*)(p);
            u.s.hi = *(const v8bf*)(p + 16);
            afrag[mt] = u.v;
        }

#pragma unroll
        for (int mt = 0; mt < 4; ++mt) {
            acc[mt][0] = __builtin_amdgcn_wmma_f32_16x16x32_bf16(
                false, afrag[mt], false, bfrag0, (short)0, acc[mt][0], false, false);
            acc[mt][1] = __builtin_amdgcn_wmma_f32_16x16x32_bf16(
                false, afrag[mt], false, bfrag1, (short)0, acc[mt][1], false, false);
        }

        // own async issues landed + all waves done reading `cur` before overwrite
        wait_async_all();
        __syncthreads();
    }

    // Epilogue: C/D layout — VGPR j: M = j (lanes 0-15) / j+8 (lanes 16-31), N = lane&15
    const int ncol = lane & 15;
    const int rsel = (lane >> 4) * 8;
#pragma unroll
    for (int mt = 0; mt < 4; ++mt) {
#pragma unroll
        for (int nt = 0; nt < 2; ++nt) {
            const int ng = n0 + nt * 16 + ncol;
            const float bv = bias[ng];
#pragma unroll
            for (int j = 0; j < 8; ++j) {
                const int row = blockM + m0w + mt * 16 + rsel + j;
                gates[(size_t)row * GSZ + ng] = acc[mt][nt][j] + bv;
            }
        }
    }
}

// ---------------------------------------------------------------------------
// LSTM cell elementwise. Writes h (bf16) into this layer's h-slot and the next
// consumer's x-slot; optionally f32 h for the decode projection.
// ---------------------------------------------------------------------------
__global__ __launch_bounds__(256) void cell_kernel(const float* __restrict__ gates,
                                                   float* __restrict__ c,
                                                   bf16* __restrict__ act_self,
                                                   bf16* __restrict__ act_next_x,
                                                   float* __restrict__ hout) {
    int idx = blockIdx.x * blockDim.x + threadIdx.x;
    if (idx >= BSZ * HID) return;
    int b = idx / HID;
    int j = idx - b * HID;
    const float* g = gates + (size_t)b * GSZ;
    float gi = sigm(g[j]);
    float gf = sigm(g[j + HID]);
    float gg = tanhf(g[j + 2 * HID]);
    float go = sigm(g[j + 3 * HID]);
    float cn = gf * c[idx] + gi * gg;
    float hn = go * tanhf(cn);
    c[idx] = cn;
    bf16 hb = f2bf(hn);
    act_self[(size_t)b * KTOT + HID + j] = hb;   // h-part for this layer's next step
    act_next_x[(size_t)b * KTOT + j] = hb;       // x-part for next consumer
    if (hout) hout[idx] = hn;
}

// vel[b, t-7, :] = hout[b,:] @ dec_W + dec_b
__global__ void decode_kernel(const float* __restrict__ hout, const float* __restrict__ decW,
                              const float* __restrict__ decb, float* __restrict__ vel, int tm7) {
    int idx = blockIdx.x * blockDim.x + threadIdx.x;
    if (idx >= BSZ * 2) return;
    int b = idx >> 1, d = idx & 1;
    const float* hr = hout + (size_t)b * HID;
    float s = 0.f;
    for (int j = 0; j < HID; ++j) s = fmaf(hr[j], decW[j * 2 + d], s);
    vel[((size_t)b * PREDN + tm7) * 2 + d] = s + decb[d];
}

// d_out = [ pred (B,12,2) | outVelocity (B,12,2) ]
__global__ void finalize_kernel(const float* __restrict__ obs, const float* __restrict__ mean,
                                const float* __restrict__ stdv, const float* __restrict__ vel,
                                float* __restrict__ out) {
    int idx = blockIdx.x * blockDim.x + threadIdx.x;
    if (idx >= BSZ * 2) return;
    int b = idx >> 1, d = idx & 1;
    float acc = obs[((size_t)b * OBSN + (OBSN - 1)) * 2 + d];
    const size_t velOutBase = (size_t)BSZ * PREDN * 2;
    for (int s = 0; s < PREDN; ++s) {
        float v = vel[((size_t)b * PREDN + s) * 2 + d];
        out[velOutBase + ((size_t)b * PREDN + s) * 2 + d] = v;
        acc = fmaf(v, stdv[d], acc + mean[d]);
        out[((size_t)b * PREDN + s) * 2 + d] = acc;
    }
}

// ---------------------------------------------------------------------------
// Host-side orchestration
// ---------------------------------------------------------------------------
extern "C" void kernel_launch(void* const* d_in, const int* in_sizes, int n_in,
                              void* d_out, int out_size, void* d_ws, size_t ws_size,
                              hipStream_t stream) {
    const float* obs    = (const float*)d_in[0];
    const float* obsVel = (const float*)d_in[1];
    const float* mean   = (const float*)d_in[2];
    const float* stdv   = (const float*)d_in[3];
    const float* encW   = (const float*)d_in[4];
    const float* encb   = (const float*)d_in[5];
    const float* decW   = (const float*)d_in[6];
    const float* decb   = (const float*)d_in[7];
    const float* Wih    = (const float*)d_in[8];
    const float* Whh    = (const float*)d_in[9];
    const float* bih    = (const float*)d_in[10];
    const float* bhh    = (const float*)d_in[11];
    float* out = (float*)d_out;

    // workspace carve-out (256B aligned slabs)
    char* w = (char*)d_ws;
    size_t off = 0;
    auto carve = [&](size_t bytes) -> void* {
        void* p = w + off;
        off = (off + bytes + 255) & ~(size_t)255;
        return p;
    };
    bf16*  wpk   = (bf16*) carve((size_t)NL * GSZ * KTOT * sizeof(bf16));   // 8 MB
    float* bias  = (float*)carve((size_t)NL * GSZ * sizeof(float));         // 16 KB
    bf16*  embed = (bf16*) carve((size_t)7 * BSZ * HID * sizeof(bf16));     // 14 MB
    bf16*  act0  = (bf16*) carve((size_t)BSZ * KTOT * sizeof(bf16));        // 4 MB
    bf16*  act1  = (bf16*) carve((size_t)BSZ * KTOT * sizeof(bf16));        // 4 MB
    float* c0    = (float*)carve((size_t)BSZ * HID * sizeof(float));        // 4 MB
    float* c1    = (float*)carve((size_t)BSZ * HID * sizeof(float));        // 4 MB
    float* gates = (float*)carve((size_t)BSZ * GSZ * sizeof(float));        // 16 MB
    float* hout  = (float*)carve((size_t)BSZ * HID * sizeof(float));        // 4 MB
    float* vel   = (float*)carve((size_t)BSZ * PREDN * 2 * sizeof(float));  // 192 KB

    const int T256 = 256;
    // --- one-time prep (recomputed every call: deterministic) ---
    pack_w_kernel<<<(NL * GSZ * KTOT + T256 - 1) / T256, T256, 0, stream>>>(Wih, Whh, wpk);
    bias_kernel<<<(NL * GSZ + T256 - 1) / T256, T256, 0, stream>>>(bih, bhh, bias);
    embed_kernel<<<(7 * BSZ * HID + T256 - 1) / T256, T256, 0, stream>>>(
        obsVel, mean, stdv, encW, encb, embed);
    // zero activations (incl. h-parts) and cell states
    zero_u32_kernel<<<(BSZ * KTOT / 2 + T256 - 1) / T256, T256, 0, stream>>>((uint32_t*)act0, BSZ * KTOT / 2);
    zero_u32_kernel<<<(BSZ * KTOT / 2 + T256 - 1) / T256, T256, 0, stream>>>((uint32_t*)act1, BSZ * KTOT / 2);
    zero_u32_kernel<<<(BSZ * HID + T256 - 1) / T256, T256, 0, stream>>>((uint32_t*)c0, BSZ * HID);
    zero_u32_kernel<<<(BSZ * HID + T256 - 1) / T256, T256, 0, stream>>>((uint32_t*)c1, BSZ * HID);

    const dim3 ggrid(GSZ / 128, BSZ / 128);   // 16 x 16 blocks
    const int cellBlocks = (BSZ * HID + T256 - 1) / T256;

    for (int t = 0; t < 19; ++t) {
        if (t < 7) {
            set_x0_kernel<<<cellBlocks, T256, 0, stream>>>(
                embed + (size_t)t * BSZ * HID, act0);
        }
        // layer 0
        gemm_gates_kernel<<<ggrid, T256, 0, stream>>>(act0, wpk, bias, gates);
        cell_kernel<<<cellBlocks, T256, 0, stream>>>(gates, c0, act0, act1, (float*)nullptr);
        // layer 1
        gemm_gates_kernel<<<ggrid, T256, 0, stream>>>(
            act1, wpk + (size_t)GSZ * KTOT, bias + GSZ, gates);
        cell_kernel<<<cellBlocks, T256, 0, stream>>>(gates, c1, act1, act0, hout);
        if (t >= 7) {
            decode_kernel<<<(BSZ * 2 + T256 - 1) / T256, T256, 0, stream>>>(
                hout, decW, decb, vel, t - 7);
        }
    }
    finalize_kernel<<<(BSZ * 2 + T256 - 1) / T256, T256, 0, stream>>>(obs, mean, stdv, vel, out);
}
